// VectorQuantizer_1795296330335
// MI455X (gfx1250) — matrix operand order repaired and checked by reading, not compiled, not run on hardware
//
#include <hip/hip_runtime.h>
#include <hip/hip_bf16.h>

#define N_Z   32768
#define N_EMB 4096
#define E_DIM 256

typedef __attribute__((ext_vector_type(16))) __bf16 v16bf;
typedef __attribute__((ext_vector_type(8)))  float  v8f;
typedef __attribute__((ext_vector_type(4)))  float  f32x4;   // clang ext-vector: OK for nontemporal builtin

union AV { v16bf v; unsigned u[8]; };

__device__ __forceinline__ unsigned short f2bf(float f) {
    unsigned u = __builtin_bit_cast(unsigned, f);
    u += 0x7FFFu + ((u >> 16) & 1u);            // round-to-nearest-even
    return (unsigned short)(u >> 16);
}
__device__ __forceinline__ float bf2f(unsigned short h) {
    unsigned u = ((unsigned)h) << 16;
    return __builtin_bit_cast(float, u);
}

// ---------------- zero-fill loss + one-hot region (streaming, NT stores) -----
__global__ void vq_zero4_kernel(f32x4* __restrict__ p, long long n4) {
    long long i = (long long)blockIdx.x * blockDim.x + threadIdx.x;
    long long stride = (long long)gridDim.x * blockDim.x;
    f32x4 zv = {0.0f, 0.0f, 0.0f, 0.0f};
    for (; i < n4; i += stride)
        __builtin_nontemporal_store(zv, &p[i]);   // don't churn L2 with 537 MB of zeros
}

// ---------------- pre-split emb into bf16 hi/lo + row norms ------------------
__global__ void vq_prep_emb_kernel(const float* __restrict__ emb,
                                   unsigned short* __restrict__ ehi,
                                   unsigned short* __restrict__ elo,
                                   float* __restrict__ e2) {
    const int n = blockIdx.x, t = threadIdx.x;
    const float v = emb[(size_t)n * E_DIM + t];
    const unsigned short h = f2bf(v);
    ehi[(size_t)n * E_DIM + t] = h;
    elo[(size_t)n * E_DIM + t] = f2bf(v - bf2f(h));
    float p = v * v;
    #pragma unroll
    for (int off = 16; off; off >>= 1) p += __shfl_xor(p, off, 32);
    __shared__ float s[8];
    if ((t & 31) == 0) s[t >> 5] = p;
    __syncthreads();
    if (t == 0) {
        float sum = 0.0f;
        #pragma unroll
        for (int w = 0; w < 8; ++w) sum += s[w];
        e2[n] = sum;
    }
}

// ---------------- verbatim emb copy into output (NT store) -------------------
__global__ void vq_copy_kernel(const float* __restrict__ src,
                               float* __restrict__ dst, int n) {
    int i = blockIdx.x * blockDim.x + threadIdx.x;
    int stride = gridDim.x * blockDim.x;
    for (; i < n; i += stride)
        __builtin_nontemporal_store(src[i], &dst[i]);
}

// ---------------- GEMM (bf16x3 split) + per-row argmin -----------------------
__global__ void __launch_bounds__(256)
vq_gemm_argmin_kernel(const float* __restrict__ z,
                      const float* __restrict__ e2,
                      const unsigned short* __restrict__ ehi,
                      const unsigned short* __restrict__ elo,
                      int* __restrict__ idx_ws,
                      float* __restrict__ idxf_out) {
    __shared__ unsigned short zh[16][E_DIM];
    __shared__ unsigned short zl[16][E_DIM];
    __shared__ float smin[8][16];
    __shared__ int   sidx[8][16];

    const int mt = blockIdx.x * 16;

    // stage 16 z rows, split fp32 -> bf16 hi + bf16 lo
    for (int t = threadIdx.x; t < 16 * E_DIM; t += 256) {
        const int r = t >> 8, c = t & 255;
        const float v = z[(size_t)(mt + r) * E_DIM + c];
        const unsigned short h = f2bf(v);
        zh[r][c] = h;
        zl[r][c] = f2bf(v - bf2f(h));
    }
    __syncthreads();

    const int wave = threadIdx.x >> 5;
    const int lane = threadIdx.x & 31;
    const int m     = lane & 15;
    const int khalf = (lane >> 4) << 3;   // A-matrix K half select (0 or 8)
    const int kb    = (lane >> 4) << 4;   // unused: B loads also use khalf (0 or 8)
    const int nloc  = lane & 15;

    // hoist the 8 A-fragments (hi and lo) into VGPRs: reused for all 32 N-tiles
    AV ah[8], al[8];
    #pragma unroll
    for (int kc = 0; kc < 8; ++kc) {
        #pragma unroll
        for (int j = 0; j < 8; ++j) {
            const int kk = (kc << 5) + ((j >> 2) << 4) + khalf + ((j & 3) << 1);
            ah[kc].u[j] = *(const unsigned*)&zh[m][kk];
            al[kc].u[j] = *(const unsigned*)&zl[m][kk];
        }
    }

    float minv[8];
    int   mini[8];
    #pragma unroll
    for (int r = 0; r < 8; ++r) { minv[r] = __builtin_inff(); mini[r] = 0x7FFFFFFF; }

    for (int nt = wave; nt < N_EMB / 16; nt += 8) {
        const int ng = nt * 16 + nloc;
        const size_t rowoff = (size_t)ng * E_DIM;

        // prefetch next tile's B rows (global_prefetch_b8) while this tile computes
        if (nt + 8 < N_EMB / 16) {
            const size_t nxt = rowoff + (size_t)8 * 16 * E_DIM;
            __builtin_prefetch(ehi + nxt, 0, 0);
            __builtin_prefetch(elo + nxt, 0, 0);
        }

        v8f acc = {};
        #pragma unroll
        for (int kc = 0; kc < 8; ++kc) {
            AV bh, bl;
            #pragma unroll
            for (int j = 0; j < 8; ++j) {
                const int kk = (kc << 5) + ((j >> 2) << 4) + khalf + ((j & 3) << 1);
                bh.u[j] = *(const unsigned*)(ehi + rowoff + kk);
                bl.u[j] = *(const unsigned*)(elo + rowoff + kk);
            }
            // z.e ~= zh.eh + zh.el + zl.eh  (error ~ lo*lo, negligible)
            acc = __builtin_amdgcn_wmma_f32_16x16x32_bf16(false, ah[kc].v, false, bh.v, (short)0, acc, false, false);
            acc = __builtin_amdgcn_wmma_f32_16x16x32_bf16(false, ah[kc].v, false, bl.v, (short)0, acc, false, false);
            acc = __builtin_amdgcn_wmma_f32_16x16x32_bf16(false, al[kc].v, false, bh.v, (short)0, acc, false, false);
        }
        // argmin key: ||e||^2 - 2 z.e   (||z||^2 is row-constant)
        const float e2n = e2[ng];
        #pragma unroll
        for (int r = 0; r < 8; ++r) {
            const float d = fmaf(-2.0f, acc[r], e2n);
            if (d < minv[r]) { minv[r] = d; mini[r] = ng; }   // strict < keeps lowest index in-lane
        }
    }

    // reduce across the 16 lanes sharing each output row (xor<16 stays in half-wave)
    #pragma unroll
    for (int off = 8; off; off >>= 1) {
        #pragma unroll
        for (int r = 0; r < 8; ++r) {
            const float ov = __shfl_xor(minv[r], off, 32);
            const int   oi = __shfl_xor(mini[r], off, 32);
            if (ov < minv[r] || (ov == minv[r] && oi < mini[r])) { minv[r] = ov; mini[r] = oi; }
        }
    }

    if (lane == 0) {
        #pragma unroll
        for (int r = 0; r < 8; ++r) { smin[wave][r] = minv[r]; sidx[wave][r] = mini[r]; }
    } else if (lane == 16) {
        #pragma unroll
        for (int r = 0; r < 8; ++r) { smin[wave][8 + r] = minv[r]; sidx[wave][8 + r] = mini[r]; }
    }
    __syncthreads();

    if (threadIdx.x < 16) {
        float bv = smin[0][threadIdx.x];
        int   bi = sidx[0][threadIdx.x];
        #pragma unroll
        for (int w = 1; w < 8; ++w) {
            const float ov = smin[w][threadIdx.x];
            const int   oi = sidx[w][threadIdx.x];
            if (ov < bv || (ov == bv && oi < bi)) { bv = ov; bi = oi; }
        }
        const int row = mt + threadIdx.x;
        idx_ws[row]   = bi;
        idxf_out[row] = (float)bi;
    }
}

// ---------------- gather z_q, one-hot scatter, per-row loss ------------------
__global__ void vq_scatter_kernel(const float* __restrict__ z,
                                  const float* __restrict__ emb,
                                  const int* __restrict__ idx_ws,
                                  float* __restrict__ onehot,
                                  float* __restrict__ zq,
                                  float* __restrict__ lossrow) {
    const int wave = threadIdx.x >> 5, lane = threadIdx.x & 31;
    const int i = blockIdx.x * 8 + wave;
    const int idx = idx_ws[i];
    const float* er = emb + (size_t)idx * E_DIM;   // L2-resident codebook row
    const float* zr = z   + (size_t)i   * E_DIM;
    float*       qr = zq  + (size_t)i   * E_DIM;
    float part = 0.0f;
    #pragma unroll
    for (int c = lane; c < E_DIM; c += 32) {
        const float e = er[c];
        __builtin_nontemporal_store(e, &qr[c]);   // z_q_st == z_q numerically; write-once stream
        const float d = e - zr[c];
        part = fmaf(d, d, part);
    }
    #pragma unroll
    for (int off = 16; off; off >>= 1) part += __shfl_xor(part, off, 32);
    if (lane == 0) {
        lossrow[i] = part;
        __builtin_nontemporal_store(1.0f, &onehot[(size_t)i * N_EMB + idx]);
    }
}

// ---------------- deterministic loss reduction -------------------------------
__global__ void vq_loss_kernel(const float* __restrict__ lossrow, float* __restrict__ out0) {
    __shared__ float s[256];
    float p = 0.0f;
    for (int i = threadIdx.x; i < N_Z; i += 256) p += lossrow[i];
    s[threadIdx.x] = p;
    __syncthreads();
    if (threadIdx.x == 0) {
        float sum = 0.0f;
        for (int t = 0; t < 256; ++t) sum += s[t];
        out0[0] = sum * (1.0f / ((float)N_Z * (float)E_DIM));
    }
}

extern "C" void kernel_launch(void* const* d_in, const int* in_sizes, int n_in,
                              void* d_out, int out_size, void* d_ws, size_t ws_size,
                              hipStream_t stream) {
    const float* z   = (const float*)d_in[0];
    const float* emb = (const float*)d_in[1];
    float* out = (float*)d_out;

    // d_out layout (flat float elements, reference return order)
    const size_t OH = 1;                                   // one-hot [N_Z, N_EMB]
    const size_t ZQ = OH + (size_t)N_Z * N_EMB;            // z_q_st  [N_Z, E_DIM]
    const size_t EM = ZQ + (size_t)N_Z * E_DIM;            // emb     [N_EMB, E_DIM]
    const size_t IX = EM + (size_t)N_EMB * E_DIM;          // indices [N_Z, 1]

    // workspace layout
    char* ws = (char*)d_ws;
    float*          e2      = (float*)ws;                                      // 16 KB
    int*            idx_ws  = (int*)(ws + 16384);                              // 128 KB
    float*          lossrow = (float*)(ws + 16384 + 131072);                   // 128 KB
    unsigned short* ehi     = (unsigned short*)(ws + 16384 + 131072 + 131072); // 2 MB
    unsigned short* elo     = ehi + (size_t)N_EMB * E_DIM;                     // 2 MB

    // 1) zero loss + one-hot region (rounded up to float4; spill overwritten later)
    const long long n4 = ((long long)1 + (long long)N_Z * N_EMB + 3) / 4;
    vq_zero4_kernel<<<4096, 256, 0, stream>>>((f32x4*)out, n4);

    // 2) split codebook to bf16 hi/lo + row norms (4 MB codebook stays L2-resident)
    vq_prep_emb_kernel<<<N_EMB, 256, 0, stream>>>(emb, ehi, elo, e2);

    // 3) emb passthrough output
    vq_copy_kernel<<<1024, 256, 0, stream>>>(emb, out + EM, N_EMB * E_DIM);

    // 4) bf16x3 WMMA GEMM + argmin
    vq_gemm_argmin_kernel<<<N_Z / 16, 256, 0, stream>>>(z, e2, ehi, elo, idx_ws, out + IX);

    // 5) gather z_q, set one-hot, per-row loss
    vq_scatter_kernel<<<N_Z / 8, 256, 0, stream>>>(z, emb, idx_ws, out + OH, out + ZQ, lossrow);

    // 6) deterministic loss mean
    vq_loss_kernel<<<1, 256, 0, stream>>>(lossrow, out);
}
